// SupConLoss_79250736546714
// MI455X (gfx1250) — 16-bit and 8-bit matrix operand order repaired and checked by reading, compiled, not run on hardware
//
#include <hip/hip_runtime.h>
#include <hip/hip_bf16.h>

typedef __attribute__((ext_vector_type(16))) _Float16 v16h;
typedef __attribute__((ext_vector_type(8)))  float    v8f;
typedef __attribute__((ext_vector_type(8)))  int      v8i;
typedef __attribute__((ext_vector_type(8)))  unsigned int v8u;

#define B_ROWS 8192
#define DIM    128
#define NCLS   100
#define TINV   (1.0f/0.3f)

__device__ __forceinline__ float grpSum(float v){
  #pragma unroll
  for (int m=1;m<16;m<<=1) v += __shfl_xor(v,m,32);
  return v;
}
__device__ __forceinline__ float grpMin(float v){
  #pragma unroll
  for (int m=1;m<16;m<<=1) v = fminf(v,__shfl_xor(v,m,32));
  return v;
}
__device__ __forceinline__ float grpMax(float v){
  #pragma unroll
  for (int m=1;m<16;m<<=1) v = fmaxf(v,__shfl_xor(v,m,32));
  return v;
}

__global__ void normalize_rows(const float* __restrict__ x,
                               float* __restrict__ outF,
                               _Float16* __restrict__ outH)
{
  const int row  = blockIdx.x;
  const int lane = threadIdx.x;
  const float4 v = ((const float4*)(x + (size_t)row*DIM))[lane];
  float ss = v.x*v.x + v.y*v.y + v.z*v.z + v.w*v.w;
  #pragma unroll
  for (int m=1;m<32;m<<=1) ss += __shfl_xor(ss,m,32);
  const float inv = 1.0f / fmaxf(sqrtf(ss), 1e-12f);
  float4 o; o.x=v.x*inv; o.y=v.y*inv; o.z=v.z*inv; o.w=v.w*inv;
  ((float4*)(outF + (size_t)row*DIM))[lane] = o;
  if (outH){
    _Float16* hp = outH + (size_t)row*DIM + lane*4;
    hp[0]=(_Float16)o.x; hp[1]=(_Float16)o.y; hp[2]=(_Float16)o.z; hp[3]=(_Float16)o.w;
  }
}

__global__ void pack_labels(const float* __restrict__ labels,
                            unsigned char* __restrict__ labU8,
                            float* __restrict__ out0)
{
  const int row = blockIdx.x, t = threadIdx.x;
  const float v = (t < NCLS) ? labels[(size_t)row*NCLS + t] : 0.0f;
  labU8[(size_t)row*DIM + t] = (v != 0.0f) ? (unsigned char)1 : (unsigned char)0;
  if (row==0 && t==0) *out0 = 0.0f;
}

__global__ void __launch_bounds__(128)
adp_kernel(const float* __restrict__ fn, const float* __restrict__ pn,
           float* __restrict__ adp)
{
  __shared__ float frow[DIM];
  const int i = blockIdx.x, t = threadIdx.x;
  frow[t] = fn[(size_t)i*DIM + t];
  __syncthreads();
  if (t < NCLS){
    const float* pr = pn + (size_t)t*DIM;
    float s = 0.0f;
    #pragma unroll 8
    for (int k=0;k<DIM;k++) s += frow[k]*pr[k];
    adp[(size_t)i*NCLS + t] = s;
  }
}

__device__ __forceinline__ void computeTile(const _Float16* __restrict__ fnh,
                                            const unsigned char* __restrict__ labU8,
                                            int j0, int hi, int ln,
                                            const v16h aF[4], const v8i aL[2],
                                            v8f& c, v8i& mi)
{
  const int rj = j0 + ln;
  const _Float16* fr = fnh + (size_t)rj*DIM;
  v8f cc = {};
  #pragma unroll
  for (int kk=0;kk<4;kk++){
    v16h b = *(const v16h*)(fr + kk*32 + hi*16);
    cc = __builtin_amdgcn_wmma_f32_16x16x32_f16(false, aF[kk], false, b,
                                                (short)0, cc, false, false);
  }
  c = cc;
  const int4* lp = (const int4*)(labU8 + (size_t)rj*DIM);
  v8i mm = {0,0,0,0,0,0,0,0};
  #pragma unroll
  for (int k2=0;k2<2;k2++){
    int4 p0 = lp[k2*4 + hi];
    int4 p1 = lp[k2*4 + 2 + hi];
    v8i b = {p0.x,p0.y,p0.z,p0.w,p1.x,p1.y,p1.z,p1.w};
    mm = __builtin_amdgcn_wmma_i32_16x16x64_iu8(false, aL[k2], false, b,
                                                mm, false, false);
  }
  mi = mm;
}

__global__ void __launch_bounds__(128)
supcon_main(const _Float16* __restrict__ fnh,
            const unsigned char* __restrict__ labU8,
            const float* __restrict__ labels,
            const float* __restrict__ adp,
            float* __restrict__ out0)
{
  const int lane = threadIdx.x & 31;
  const int wave = threadIdx.x >> 5;
  const int hi   = lane >> 4;
  const int ln   = lane & 15;
  const int i0   = (blockIdx.x*4 + wave) * 16;
  const int ri   = i0 + ln;

  v16h aF[4];
  {
    const unsigned int* frp = (const unsigned int*)(fnh + (size_t)ri*DIM);
    #pragma unroll
    for (int kk=0;kk<4;kk++){
      v8u tmp;
      #pragma unroll
      for (int r=0;r<8;r++)
        tmp[r] = frp[kk*16 + hi*8 + r];
      aF[kk] = __builtin_bit_cast(v16h, tmp);
    }
  }
  v8i aL[2];
  {
    const unsigned int* lrp = (const unsigned int*)(labU8 + (size_t)ri*DIM);
    #pragma unroll
    for (int k2=0;k2<2;k2++){
      v8i t;
      #pragma unroll
      for (int v=0;v<8;v++)
        t[v] = (int)lrp[k2*16 + (v>>2)*8 + hi*4 + (v&3)];
      aL[k2] = t;
    }
  }

  float pmin[8], pmax[8];
  #pragma unroll
  for (int r=0;r<8;r++){ pmin[r] = __builtin_inff(); pmax[r] = -__builtin_inff(); }

  for (int j0=0; j0<B_ROWS; j0+=16){
    v8f c; v8i mi;
    computeTile(fnh, labU8, j0, hi, ln, aF, aL, c, mi);
    #pragma unroll
    for (int r=0;r<8;r++){
      const bool pos = mi[r] > 0;
      const float cv = c[r];
      pmin[r] = fminf(pmin[r], pos ? cv : 0.0f);
      pmax[r] = fmaxf(pmax[r], pos ? 0.0f : cv);
    }
  }
  #pragma unroll
  for (int r=0;r<8;r++){ pmin[r] = grpMin(pmin[r]); pmax[r] = grpMax(pmax[r]); }

  float sP[8], sN[8], cP[8];
  #pragma unroll
  for (int r=0;r<8;r++){ sP[r]=0.0f; sN[r]=0.0f; cP[r]=0.0f; }

  for (int j0=0; j0<B_ROWS; j0+=16){
    v8f c; v8i mi;
    computeTile(fnh, labU8, j0, hi, ln, aF, aL, c, mi);
    #pragma unroll
    for (int r=0;r<8;r++){
      const bool pos = mi[r] > 0;
      const float cv = c[r];
      const bool zc = pos ? (cv - 0.5f >= pmax[r]) : (cv + 0.5f <= pmin[r]);
      const float cmod = zc ? 0.0f : cv;
      const float e = __expf(cv * TINV);
      sP[r] += pos ? e * __expf((-1.0f - cmod)*0.25f) : 0.0f;
      sN[r] += pos ? 0.0f : e * __expf(-1.0f + cmod);
      cP[r] += pos ? 1.0f : 0.0f;
    }
  }
  #pragma unroll
  for (int r=0;r<8;r++){ sP[r]=grpSum(sP[r]); sN[r]=grpSum(sN[r]); cP[r]=grpSum(cP[r]); }

  if (ln < 8){
    float sumP=0.0f, sumN=0.0f, cnt=0.0f;
    #pragma unroll
    for (int r=0;r<8;r++) if (ln==r){ sumP=sP[r]; sumN=sN[r]; cnt=cP[r]; }
    const int row = i0 + hi*8 + ln;
    const float* lrow = labels + (size_t)row*NCLS;
    const float* arow = adp    + (size_t)row*NCLS;
    float pos2=0.0f, neg2=0.0f, lsum=0.0f;
    for (int ci=0; ci<NCLS; ci++){
      const float ap = arow[ci], lb = lrow[ci];
      const float e2 = __expf(ap * TINV);
      pos2 += lb * e2 * __expf((-1.0f - ap)*0.25f);
      neg2 += (1.0f - lb) * e2 * __expf(-1.0f + ap);
      lsum += lb;
    }
    const float lamP = cnt / lsum;
    const float lamN = ((float)B_ROWS - cnt) / ((float)NCLS - lsum);
    const float num = sumP + lamP * pos2;
    const float den = sumN + lamN * neg2 + num;
    const float loss = -__logf(num / den);
    atomicAdd(out0, loss * (1.0f / (float)B_ROWS));
  }
}

extern "C" void kernel_launch(void* const* d_in, const int* in_sizes, int n_in,
                              void* d_out, int out_size, void* d_ws, size_t ws_size,
                              hipStream_t stream)
{
  const float* features   = (const float*)d_in[0];
  const float* prototypes = (const float*)d_in[1];
  const float* labels     = (const float*)d_in[2];
  float* out = (float*)d_out;

  char* ws = (char*)d_ws;
  size_t off = 0;
  float* fn = (float*)(ws + off);            off += (size_t)B_ROWS*DIM*sizeof(float);
  float* pn = (float*)(ws + off);            off += 64*1024;
  _Float16* fnh = (_Float16*)(ws + off);     off += (size_t)B_ROWS*DIM*sizeof(_Float16);
  _Float16* pnh = (_Float16*)(ws + off);     off += 64*1024;
  unsigned char* labU8 = (unsigned char*)(ws + off);

  normalize_rows<<<B_ROWS, 32, 0, stream>>>(features, fn, fnh);
  normalize_rows<<<NCLS,   32, 0, stream>>>(prototypes, pn, pnh);
  pack_labels  <<<B_ROWS, 128, 0, stream>>>(labels, labU8, out);
  adp_kernel   <<<B_ROWS, 128, 0, stream>>>(fn, pn, out + 1);
  supcon_main  <<<B_ROWS/64, 128, 0, stream>>>(fnh, labU8, labels, out + 1, out);
}
